// MoDLayer_85031762526475
// MI455X (gfx1250) — matrix operand order repaired and checked by reading, not compiled, not run on hardware
//
#include <hip/hip_runtime.h>
#include <cmath>

// ---------------------------------------------------------------------------
// MoD layer for MI455X (gfx1250): all GEMMs via v_wmma_f32_16x16x32_bf16,
// wave32, async global->LDS tile staging (GLOBAL_LOAD_ASYNC_TO_LDS_B128 +
// s_wait_asynccnt), global_prefetch_b8 for next tiles, bf16 operands
// everywhere (weights pre-converted + pre-transposed to K-contiguous layout).
// ---------------------------------------------------------------------------

#define TPB 256

constexpr int BATCH = 4;
constexpr int T     = 2048;
constexpr int D     = 2048;
constexpr int NH    = 16;
constexpr int HD    = 128;       // D / NH
constexpr int KSEL  = 1024;      // T * CAPACITY
constexpr int DFF   = 8192;
constexpr float EPSR = 1e-6f;

typedef __bf16 bf16;
typedef __attribute__((ext_vector_type(16))) __bf16 bf16x16;
typedef __attribute__((ext_vector_type(8)))  float  f32x8;
typedef __attribute__((ext_vector_type(4)))  int    i32x4;

// --- CDNA5 async global->LDS staging (guarded: falls back to sync copy) ----
#if defined(__has_builtin)
#if __has_builtin(__builtin_amdgcn_global_load_async_to_lds_b128) && \
    __has_builtin(__builtin_amdgcn_s_wait_asynccnt)
#define ASYNC_LDS 1
#endif
#endif
#ifndef ASYNC_LDS
#define ASYNC_LDS 0
#endif

__device__ __forceinline__ void cp16(bf16* ldsDst, const bf16* gSrc) {
#if ASYNC_LDS
  // signature: (v4i addrspace(1)*, v4i addrspace(3)*, imm offset, imm cpol)
  __builtin_amdgcn_global_load_async_to_lds_b128(
      (__attribute__((address_space(1))) i32x4*)(void*)gSrc,
      (__attribute__((address_space(3))) i32x4*)(void*)ldsDst, 0, 0);
#else
  *(uint4*)ldsDst = *(const uint4*)gSrc;
#endif
}

__device__ __forceinline__ void wait_cp() {
#if ASYNC_LDS
  __builtin_amdgcn_s_wait_asynccnt(0);
#endif
}

// ---------------------------------------------------------------------------
// 0) Weight convert + transpose: W[K][N] fp32 -> WT[N][K] bf16 (K contiguous)
// ---------------------------------------------------------------------------
__global__ void convT_kernel(const float* __restrict__ W, bf16* __restrict__ WT,
                             int Kdim, int logN) {
  long long t = (long long)blockIdx.x * blockDim.x + threadIdx.x;  // over K*N
  long long n = t & ((1LL << logN) - 1);
  long long k = t >> logN;
  WT[n * Kdim + k] = (bf16)W[t];
}

// ---------------------------------------------------------------------------
// 1) Router: logits = x . w_router, causal_logits = x . w_causal
// ---------------------------------------------------------------------------
__global__ void router_kernel(const float* __restrict__ hs,
                              const float* __restrict__ wr,
                              const float* __restrict__ wc,
                              float* __restrict__ L, float* __restrict__ CL) {
  __shared__ float s1[TPB], s2[TPB];
  const int row = blockIdx.x;                       // b*T + t
  const float* x = hs + (size_t)row * D;
  float a = 0.f, b = 0.f;
  for (int d = threadIdx.x; d < D; d += TPB) { float v = x[d]; a += v * wr[d]; b += v * wc[d]; }
  s1[threadIdx.x] = a; s2[threadIdx.x] = b; __syncthreads();
  for (int s = TPB / 2; s > 0; s >>= 1) {
    if (threadIdx.x < s) { s1[threadIdx.x] += s1[threadIdx.x + s]; s2[threadIdx.x] += s2[threadIdx.x + s]; }
    __syncthreads();
  }
  if (threadIdx.x == 0) { L[row] = s1[0]; CL[row] = s2[0]; }
}

// ---------------------------------------------------------------------------
// 2) Per-batch exact top-k via bitonic sort; then sort selection by position.
// ---------------------------------------------------------------------------
__global__ __launch_bounds__(1024)
void topk_kernel(const float* __restrict__ L, int* __restrict__ idx,
                 float* __restrict__ gate, float* __restrict__ mask) {
  __shared__ float sv[T];
  __shared__ int   si[T];
  const int b = blockIdx.x, tid = threadIdx.x;
  for (int i = tid; i < T; i += 1024) { sv[i] = L[b * T + i]; si[i] = i; mask[b * T + i] = 0.f; }
  __syncthreads();
  // descending by value, ties -> lower index first (matches lax.top_k)
  for (int size = 2; size <= T; size <<= 1)
    for (int stride = size >> 1; stride > 0; stride >>= 1) {
      for (int base = 0; base < T; base += 1024) {
        int i = base + tid, j = i ^ stride;
        if (j > i) {
          float av = sv[i], bw = sv[j]; int ai = si[i], bi = si[j];
          bool before = (av > bw) || (av == bw && ai < bi);
          bool up = ((i & size) == 0);
          if (up ? !before : before) { sv[i] = bw; sv[j] = av; si[i] = bi; si[j] = ai; }
        }
      }
      __syncthreads();
    }
  // sort the kept KSEL entries ascending by original position
  for (int size = 2; size <= KSEL; size <<= 1)
    for (int stride = size >> 1; stride > 0; stride >>= 1) {
      int i = tid, j = i ^ stride;
      if (j > i && j < KSEL) {
        int ai = si[i], bi = si[j];
        bool before = ai < bi;
        bool up = ((i & size) == 0);
        if (up ? !before : before) {
          float tv = sv[i]; sv[i] = sv[j]; sv[j] = tv;
          si[i] = bi; si[j] = ai;
        }
      }
      __syncthreads();
    }
  if (tid < KSEL) {
    idx[b * KSEL + tid]  = si[tid];
    gate[b * KSEL + tid] = 1.f / (1.f + __expf(-sv[tid]));
    mask[b * T + si[tid]] = 1.f;
  }
}

// ---------------------------------------------------------------------------
// 3) Losses (deterministic two-stage reduction)
// ---------------------------------------------------------------------------
__global__ void loss_partial_kernel(const float* __restrict__ L, const float* __restrict__ CL,
                                    const float* __restrict__ MK, float* __restrict__ part) {
  __shared__ float r0[TPB], r1[TPB], r2[TPB];
  int i = blockIdx.x * TPB + threadIdx.x;
  float l = L[i], y = MK[i], c = CL[i];
  r0[threadIdx.x] = fmaxf(l, 0.f) - l * y + log1pf(__expf(-fabsf(l)));
  r1[threadIdx.x] = l * l;
  r2[threadIdx.x] = fmaxf(c, 0.f) - c * y + log1pf(__expf(-fabsf(c)));
  __syncthreads();
  for (int s = TPB / 2; s > 0; s >>= 1) {
    if (threadIdx.x < s) {
      r0[threadIdx.x] += r0[threadIdx.x + s];
      r1[threadIdx.x] += r1[threadIdx.x + s];
      r2[threadIdx.x] += r2[threadIdx.x + s];
    }
    __syncthreads();
  }
  if (threadIdx.x == 0) {
    part[blockIdx.x * 3 + 0] = r0[0];
    part[blockIdx.x * 3 + 1] = r1[0];
    part[blockIdx.x * 3 + 2] = r2[0];
  }
}

__global__ void loss_final_kernel(const float* __restrict__ part, float* __restrict__ out, int npart) {
  if (blockIdx.x == 0 && threadIdx.x == 0) {
    float s0 = 0.f, s1 = 0.f, s2 = 0.f;
    for (int p = 0; p < npart; ++p) { s0 += part[p * 3]; s1 += part[p * 3 + 1]; s2 += part[p * 3 + 2]; }
    const float inv = 1.f / (float)(BATCH * T);
    out[0] = s0 * inv;          // router BCE
    out[1] = s1 * inv * 1e-4f;  // z-loss
    out[2] = s2 * inv;          // causal BCE
  }
}

// ---------------------------------------------------------------------------
// 4) (Gather +) RMSNorm -> bf16 A-matrix (+ optional fp32 copy of selection)
// ---------------------------------------------------------------------------
__global__ void rms_kernel(const float* __restrict__ src, const int* __restrict__ idx,
                           const float* __restrict__ lnw, bf16* __restrict__ outb,
                           float* __restrict__ selOut) {
  __shared__ float red[TPB];
  const int r = blockIdx.x;                         // [0, BATCH*KSEL)
  const float* x;
  if (idx) { int b = r / KSEL; x = src + ((size_t)b * T + idx[r]) * D; }
  else     { x = src + (size_t)r * D; }
  float ss = 0.f;
  for (int d = threadIdx.x; d < D; d += TPB) { float v = x[d]; ss += v * v; }
  red[threadIdx.x] = ss; __syncthreads();
  for (int s = TPB / 2; s > 0; s >>= 1) {
    if (threadIdx.x < s) red[threadIdx.x] += red[threadIdx.x + s];
    __syncthreads();
  }
  const float rms = rsqrtf(red[0] / (float)D + EPSR);
  for (int d = threadIdx.x; d < D; d += TPB) {
    float v = x[d];
    outb[(size_t)r * D + d] = (bf16)(v * rms * lnw[d]);
    if (selOut) selOut[(size_t)r * D + d] = v;
  }
}

// ---------------------------------------------------------------------------
// 5) bf16 WMMA GEMM: C[z][M,N] = A[z][M,K] * B[z][K,N] (+bias)(+residual)
//    A bf16: row-major, K contiguous (lda = row stride).
//    B bf16: K contiguous, ldbN = stride between N columns.
//    Block tile 128x128x32; 8 waves, each computes a 32x64 patch = 8 WMMA.
//    Tiles staged to LDS with GLOBAL_LOAD_ASYNC_TO_LDS_B128.
// ---------------------------------------------------------------------------
#define BM 128
#define BN 128
#define BKD 32

__global__ __launch_bounds__(256)
void gemm_bf16_kernel(const bf16* __restrict__ A, int lda, long long za0, long long za1,
                      const bf16* __restrict__ Bm, int ldbN, long long zb0, long long zb1,
                      const float* __restrict__ bias,
                      const float* __restrict__ resid,
                      float* __restrict__ Cf, bf16* __restrict__ Cb,
                      int ldc, long long zc0, long long zc1,
                      int M, int N, int Kdim, int zH) {
  __shared__ bf16 As[BM * BKD];
  __shared__ bf16 Bs[BN * BKD];   // Bs[n][k] (K contiguous)
  const int tid = threadIdx.x;
  const int lane = tid & 31, wid = tid >> 5;
  const int wm = wid >> 1, wn = wid & 1;            // 4 x 2 wave grid
  const int mBlk = blockIdx.y * BM, nBlk = blockIdx.x * BN;
  const int bz = blockIdx.z / zH, hz = blockIdx.z % zH;
  const bf16* Abase = A  + (long long)bz * za0 + (long long)hz * za1;
  const bf16* Bbase = Bm + (long long)bz * zb0 + (long long)hz * zb1;
  const long long offC = (long long)bz * zc0 + (long long)hz * zc1;

  f32x8 acc[2][4];
#pragma unroll
  for (int i1 = 0; i1 < 2; ++i1)
#pragma unroll
    for (int i2 = 0; i2 < 4; ++i2)
#pragma unroll
      for (int e = 0; e < 8; ++e) acc[i1][i2][e] = 0.f;

  for (int k0 = 0; k0 < Kdim; k0 += BKD) {
    // --- stage tiles: 16B chunks, K-contiguous, async to LDS ---
#pragma unroll
    for (int c = tid; c < (BM * BKD) / 8; c += 256) {
      int m = c >> 2, kg = (c & 3) << 3;
      cp16(&As[m * BKD + kg], Abase + (long long)(mBlk + m) * lda + (k0 + kg));
    }
#pragma unroll
    for (int c = tid; c < (BN * BKD) / 8; c += 256) {
      int n = c >> 2, kg = (c & 3) << 3;
      cp16(&Bs[n * BKD + kg], Bbase + (long long)(nBlk + n) * ldbN + (k0 + kg));
    }
    wait_cp();
    __syncthreads();

    // prefetch next K tiles into cache (global_prefetch_b8)
    if (k0 + BKD < Kdim) {
      __builtin_prefetch(Abase + (long long)(mBlk + (tid >> 1)) * lda + (k0 + BKD), 0, 1);
      __builtin_prefetch(Bbase + (long long)(nBlk + (tid >> 1)) * ldbN + (k0 + BKD), 0, 1);
    }

    // --- fragments: contiguous k order (k = 16*(lane/16) + e) in both A and B ---
    bf16x16 af[2], bfm[4];
#pragma unroll
    for (int tm = 0; tm < 2; ++tm) {
      const int m0 = wm * 32 + tm * 16 + (lane & 15);
      const bf16* rowp = &As[m0 * BKD];
      const int half8 = (lane >> 4) << 4;
#pragma unroll
      for (int v2 = 0; v2 < 8; ++v2) {     // A 16x32: lane<16 K=0..15, lane>=16 K=16..31
        int e = 2 * v2;
        int kk = e + half8;
        af[tm][e]     = rowp[kk];
        af[tm][e + 1] = rowp[kk + 1];
      }
    }
#pragma unroll
    for (int tn = 0; tn < 4; ++tn) {       // B 32x16: lane<16 K=0..15, lane>=16 K=16..31
      const int n0 = wn * 64 + tn * 16 + (lane & 15);
      const bf16* colp = &Bs[n0 * BKD + ((lane >> 4) << 4)];
#pragma unroll
      for (int e = 0; e < 16; ++e) bfm[tn][e] = colp[e];
    }
#pragma unroll
    for (int tm = 0; tm < 2; ++tm)
#pragma unroll
      for (int tn = 0; tn < 4; ++tn)
        acc[tm][tn] = __builtin_amdgcn_wmma_f32_16x16x32_bf16(
            false, af[tm], false, bfm[tn], (short)0, acc[tm][tn], false, false);
    __syncthreads();
  }

  // --- epilogue: C layout -> row = r + 8*(lane>=16), col = lane&15 ---
  const int colL = lane & 15, halfL = lane >> 4;
#pragma unroll
  for (int tm = 0; tm < 2; ++tm)
#pragma unroll
    for (int tn = 0; tn < 4; ++tn) {
      int mg = mBlk + wm * 32 + tm * 16;
      int ng = nBlk + wn * 64 + tn * 16 + colL;
#pragma unroll
      for (int rr = 0; rr < 8; ++rr) {
        int row = mg + rr + 8 * halfL;
        if (row < M && ng < N) {
          float v = acc[tm][tn][rr];
          if (bias)  v += bias[ng];
          long long ci = offC + (long long)row * ldc + ng;
          if (resid) v += resid[ci];
          if (Cf) Cf[ci] = v;
          if (Cb) Cb[ci] = (bf16)v;
        }
      }
    }
}

// ---------------------------------------------------------------------------
// 6) RoPE on fp32 q/k -> bf16 (positions are the selected token indices)
// ---------------------------------------------------------------------------
__global__ void rope_kernel(const float* __restrict__ src, const int* __restrict__ idx,
                            bf16* __restrict__ dst) {
  int t = blockIdx.x * blockDim.x + threadIdx.x;   // BATCH*KSEL*NH*(HD/2)
  int d2 = t & 63;
  int h  = (t >> 6) & (NH - 1);
  int i  = (t >> 10) & (KSEL - 1);
  int b  = t >> 20;
  float pos  = (float)idx[b * KSEL + i];
  float freq = __powf(10000.f, -((float)(2 * d2)) / (float)HD);
  float ang  = pos * freq;
  float s, c;
  __sincosf(ang, &s, &c);
  size_t base = (((size_t)(b * KSEL + i)) * NH + h) * HD;
  float x1 = src[base + d2], x2 = src[base + 64 + d2];
  dst[base + d2]      = (bf16)(x1 * c - x2 * s);
  dst[base + 64 + d2] = (bf16)(x2 * c + x1 * s);
}

// ---------------------------------------------------------------------------
// 6b) V transpose: VB[B,k,H,hd] -> VT[B,H,hd,k]  (k becomes contiguous)
// ---------------------------------------------------------------------------
__global__ void vT_kernel(const bf16* __restrict__ VB, bf16* __restrict__ VT) {
  long long t = (long long)blockIdx.x * blockDim.x + threadIdx.x;  // BATCH*KSEL*NH*HD
  int d = (int)(t & (HD - 1));
  int h = (int)((t >> 7) & (NH - 1));
  int j = (int)((t >> 11) & (KSEL - 1));
  int b = (int)(t >> 21);
  VT[(((long long)(b * NH + h) * HD) + d) * KSEL + j] = VB[t];
}

// ---------------------------------------------------------------------------
// 7) Causal masked softmax over a k-row; writes bf16 probabilities
// ---------------------------------------------------------------------------
__global__ void softmax_kernel(const float* __restrict__ SC, bf16* __restrict__ P) {
  __shared__ float red[TPB];
  const int rid = blockIdx.x;                 // (b*NH + h)*KSEL + i
  const int i = rid & (KSEL - 1);
  const float* srow = SC + (size_t)rid * KSEL;
  bf16* prow = P + (size_t)rid * KSEL;
  const float scale = 0.08838834764831845f;   // 1/sqrt(128)
  float m = -3.4e38f;
  for (int j = threadIdx.x; j < KSEL; j += TPB) if (j <= i) m = fmaxf(m, srow[j] * scale);
  red[threadIdx.x] = m; __syncthreads();
  for (int s = TPB / 2; s > 0; s >>= 1) {
    if (threadIdx.x < s) red[threadIdx.x] = fmaxf(red[threadIdx.x], red[threadIdx.x + s]);
    __syncthreads();
  }
  m = red[0]; __syncthreads();
  float sum = 0.f;
  for (int j = threadIdx.x; j < KSEL; j += TPB) if (j <= i) sum += __expf(srow[j] * scale - m);
  red[threadIdx.x] = sum; __syncthreads();
  for (int s = TPB / 2; s > 0; s >>= 1) {
    if (threadIdx.x < s) red[threadIdx.x] += red[threadIdx.x + s];
    __syncthreads();
  }
  const float inv = 1.f / red[0];
  for (int j = threadIdx.x; j < KSEL; j += TPB)
    prow[j] = (bf16)((j <= i) ? __expf(srow[j] * scale - m) * inv : 0.f);
}

// ---------------------------------------------------------------------------
// 8) SwiGLU fuse: ffb = bf16(silu(gate) * up)
// ---------------------------------------------------------------------------
__global__ void silu_mul_kernel(const float* __restrict__ G, const float* __restrict__ U,
                                bf16* __restrict__ O, long long n) {
  long long t = (long long)blockIdx.x * blockDim.x + threadIdx.x;
  if (t < n) { float g = G[t]; O[t] = (bf16)(g / (1.f + __expf(-g)) * U[t]); }
}

// ---------------------------------------------------------------------------
// 9) Soft-gated scatter back into full hidden states
// ---------------------------------------------------------------------------
__global__ void scatter_kernel(const float* __restrict__ SEL, const float* __restrict__ BLK,
                               const float* __restrict__ gate, const int* __restrict__ idx,
                               float* __restrict__ out) {
  long long t = (long long)blockIdx.x * blockDim.x + threadIdx.x;  // BATCH*KSEL*D
  int d = (int)(t & 2047);
  int i = (int)((t >> 11) & 1023);
  int b = (int)(t >> 21);
  float sv = SEL[t], bv2 = BLK[t];
  float g = gate[b * KSEL + i];
  int tp = idx[b * KSEL + i];
  out[(((size_t)b * T) + tp) * D + d] = sv + g * (bv2 - sv);
}

// ---------------------------------------------------------------------------
// Host-side orchestration
// ---------------------------------------------------------------------------
extern "C" void kernel_launch(void* const* d_in, const int* in_sizes, int n_in,
                              void* d_out, int out_size, void* d_ws, size_t ws_size,
                              hipStream_t stream) {
  const float* hs  = (const float*)d_in[0];
  const float* wr  = (const float*)d_in[1];
  const float* wc  = (const float*)d_in[2];
  const float* ln1 = (const float*)d_in[3];
  const float* ln2 = (const float*)d_in[4];
  const float* wq  = (const float*)d_in[5];
  const float* bq  = (const float*)d_in[6];
  const float* wk  = (const float*)d_in[7];
  const float* bk  = (const float*)d_in[8];
  const float* wv  = (const float*)d_in[9];
  const float* bv  = (const float*)d_in[10];
  const float* wo  = (const float*)d_in[11];
  const float* wg  = (const float*)d_in[12];
  const float* wu  = (const float*)d_in[13];
  const float* wd  = (const float*)d_in[14];
  float* out = (float*)d_out;

  char* ws = (char*)d_ws;
  size_t off = 0;
  auto alloc = [&](size_t bytes) { size_t o = off; off = (off + bytes + 255) & ~(size_t)255; return o; };

  const size_t M_ROWS = (size_t)BATCH * KSEL;          // 4096
  float* L    = (float*)(ws + alloc((size_t)BATCH * T * 4));
  float* CL   = (float*)(ws + alloc((size_t)BATCH * T * 4));
  float* MK   = (float*)(ws + alloc((size_t)BATCH * T * 4));
  int*   IDX  = (int*)  (ws + alloc(M_ROWS * 4));
  float* GATE = (float*)(ws + alloc(M_ROWS * 4));
  float* PART = (float*)(ws + alloc(32 * 3 * 4));
  // bf16, transposed weights (K contiguous)
  bf16* WQT = (bf16*)(ws + alloc((size_t)D * D * 2));
  bf16* WKT = (bf16*)(ws + alloc((size_t)D * D * 2));
  bf16* WVT = (bf16*)(ws + alloc((size_t)D * D * 2));
  bf16* WOT = (bf16*)(ws + alloc((size_t)D * D * 2));
  bf16* WGT = (bf16*)(ws + alloc((size_t)D * DFF * 2));
  bf16* WUT = (bf16*)(ws + alloc((size_t)D * DFF * 2));
  bf16* WDT = (bf16*)(ws + alloc((size_t)DFF * D * 2));
  // activations
  float* SEL  = (float*)(ws + alloc(M_ROWS * D * 4));
  bf16*  HA   = (bf16*) (ws + alloc(M_ROWS * D * 2));
  float* Qf   = (float*)(ws + alloc(M_ROWS * D * 4));
  float* Kf   = (float*)(ws + alloc(M_ROWS * D * 4));
  bf16*  VB   = (bf16*) (ws + alloc(M_ROWS * D * 2));
  bf16*  VT   = (bf16*) (ws + alloc(M_ROWS * D * 2));
  bf16*  QB   = (bf16*) (ws + alloc(M_ROWS * D * 2));
  bf16*  KB   = (bf16*) (ws + alloc(M_ROWS * D * 2));
  float* SC   = (float*)(ws + alloc((size_t)BATCH * NH * KSEL * KSEL * 4));
  bf16*  P    = (bf16*) (ws + alloc((size_t)BATCH * NH * KSEL * KSEL * 2));
  bf16*  CTXB = (bf16*) (ws + alloc(M_ROWS * D * 2));
  float* X    = (float*)(ws + alloc(M_ROWS * D * 4));
  bf16*  H2B  = (bf16*) (ws + alloc(M_ROWS * D * 2));
  // region reuse (dead by the time these are written):
  float* GT  = SC;                    // [4096, DFF] fp32
  float* UP  = SC + M_ROWS * DFF;     // [4096, DFF] fp32
  bf16*  FFB = (bf16*)Qf;             // 67MB over dead Qf+Kf (contiguous)
  float* BLK = (float*)P;             // block output over dead P
  (void)in_sizes; (void)n_in; (void)out_size; (void)ws_size;

  // new_states starts as a copy of hidden_states
  (void)hipMemcpyAsync(out, hs, (size_t)BATCH * T * D * 4, hipMemcpyDeviceToDevice, stream);

  // weights -> bf16 transposed
  {
    long long nDD = (long long)D * D, nDF = (long long)D * DFF;
    convT_kernel<<<(int)(nDD / TPB), TPB, 0, stream>>>(wq, WQT, D, 11);
    convT_kernel<<<(int)(nDD / TPB), TPB, 0, stream>>>(wk, WKT, D, 11);
    convT_kernel<<<(int)(nDD / TPB), TPB, 0, stream>>>(wv, WVT, D, 11);
    convT_kernel<<<(int)(nDD / TPB), TPB, 0, stream>>>(wo, WOT, D, 11);
    convT_kernel<<<(int)(nDF / TPB), TPB, 0, stream>>>(wg, WGT, D, 13);
    convT_kernel<<<(int)(nDF / TPB), TPB, 0, stream>>>(wu, WUT, D, 13);
    convT_kernel<<<(int)(nDF / TPB), TPB, 0, stream>>>(wd, WDT, DFF, 11);
  }

  router_kernel<<<BATCH * T, TPB, 0, stream>>>(hs, wr, wc, L, CL);
  topk_kernel<<<BATCH, 1024, 0, stream>>>(L, IDX, GATE, MK);
  loss_partial_kernel<<<32, TPB, 0, stream>>>(L, CL, MK, PART);
  loss_final_kernel<<<1, 32, 0, stream>>>(PART, out + (size_t)BATCH * T * D, 32);

  // gather + RMSNorm1 -> SEL (fp32), HA (bf16)
  rms_kernel<<<(int)M_ROWS, TPB, 0, stream>>>(hs, IDX, ln1, HA, SEL);

  dim3 blk(256);
  const long long kHD = (long long)KSEL * NH * HD;     // per-b stride of [B,k,H,hd]
  // --- QKV projections ---
  {
    dim3 g(D / BN, (int)M_ROWS / BM, 1);
    gemm_bf16_kernel<<<g, blk, 0, stream>>>(HA, D, 0, 0, WQT, D, 0, 0,
                                            bq, nullptr, Qf, nullptr, D, 0, 0,
                                            (int)M_ROWS, D, D, 1);
    gemm_bf16_kernel<<<g, blk, 0, stream>>>(HA, D, 0, 0, WKT, D, 0, 0,
                                            bk, nullptr, Kf, nullptr, D, 0, 0,
                                            (int)M_ROWS, D, D, 1);
    gemm_bf16_kernel<<<g, blk, 0, stream>>>(HA, D, 0, 0, WVT, D, 0, 0,
                                            bv, nullptr, nullptr, VB, D, 0, 0,
                                            (int)M_ROWS, D, D, 1);
  }
  // --- RoPE -> bf16 q,k ; V -> [B,H,hd,k] ---
  {
    int total = BATCH * KSEL * NH * (HD / 2);
    rope_kernel<<<total / TPB, TPB, 0, stream>>>(Qf, IDX, QB);
    rope_kernel<<<total / TPB, TPB, 0, stream>>>(Kf, IDX, KB);
    long long n = (long long)M_ROWS * D;
    vT_kernel<<<(int)(n / TPB), TPB, 0, stream>>>(VB, VT);
  }
  // --- scores[b,h] = q (k x hd) x k^T (hd x k) ---
  {
    dim3 g(KSEL / BN, KSEL / BM, BATCH * NH);
    gemm_bf16_kernel<<<g, blk, 0, stream>>>(
        QB, NH * HD, kHD, HD,                   // A: q rows, hd contiguous
        KB, NH * HD, kHD, HD,                   // B: columns j, hd contiguous
        nullptr, nullptr, SC, nullptr,
        KSEL, (long long)NH * KSEL * KSEL, (long long)KSEL * KSEL,
        KSEL, KSEL, HD, NH);
  }
  softmax_kernel<<<BATCH * NH * KSEL, TPB, 0, stream>>>(SC, P);
  // --- ctx[b,:,h,:] = P (k x k) x V (k x hd) via VT (k contiguous) ---
  {
    dim3 g(HD / BN, KSEL / BM, BATCH * NH);
    gemm_bf16_kernel<<<g, blk, 0, stream>>>(
        P, KSEL, (long long)NH * KSEL * KSEL, (long long)KSEL * KSEL,
        VT, KSEL, (long long)NH * HD * KSEL, (long long)HD * KSEL,
        nullptr, nullptr, nullptr, CTXB,
        NH * HD, kHD, HD,
        KSEL, HD, KSEL, NH);
  }
  // --- x = sel + ctx @ wo ---
  {
    dim3 g(D / BN, (int)M_ROWS / BM, 1);
    gemm_bf16_kernel<<<g, blk, 0, stream>>>(CTXB, D, 0, 0, WOT, D, 0, 0,
                                            nullptr, SEL, X, nullptr, D, 0, 0,
                                            (int)M_ROWS, D, D, 1);
  }
  // --- RMSNorm2 -> H2B ---
  rms_kernel<<<(int)M_ROWS, TPB, 0, stream>>>(X, nullptr, ln2, H2B, nullptr);
  // --- MLP ---
  {
    dim3 g(DFF / BN, (int)M_ROWS / BM, 1);
    gemm_bf16_kernel<<<g, blk, 0, stream>>>(H2B, D, 0, 0, WGT, D, 0, 0,
                                            nullptr, nullptr, GT, nullptr, DFF, 0, 0,
                                            (int)M_ROWS, DFF, D, 1);
    gemm_bf16_kernel<<<g, blk, 0, stream>>>(H2B, D, 0, 0, WUT, D, 0, 0,
                                            nullptr, nullptr, UP, nullptr, DFF, 0, 0,
                                            (int)M_ROWS, DFF, D, 1);
  }
  {
    long long n = (long long)M_ROWS * DFF;
    silu_mul_kernel<<<(int)(n / TPB), TPB, 0, stream>>>(GT, UP, FFB, n);
  }
  {
    dim3 g(D / BN, (int)M_ROWS / BM, 1);
    gemm_bf16_kernel<<<g, blk, 0, stream>>>(FFB, DFF, 0, 0, WDT, DFF, 0, 0,
                                            nullptr, X, BLK, nullptr, D, 0, 0,
                                            (int)M_ROWS, D, DFF, 1);
  }
  // --- gated scatter into d_out ---
  {
    long long n = (long long)M_ROWS * D;
    scatter_kernel<<<(int)(n / TPB), TPB, 0, stream>>>(SEL, BLK, GATE, IDX, out);
  }
}
